// GraphAttention_12189117186450
// MI455X (gfx1250) — matrix operand order repaired and checked by reading, not compiled, not run on hardware
//
#include <hip/hip_runtime.h>
#include <hip/hip_bf16.h>
#include <hip/hip_fp16.h>

typedef __attribute__((ext_vector_type(16))) _Float16 v16h;
typedef __attribute__((ext_vector_type(8)))  float    v8f;
typedef __attribute__((ext_vector_type(2)))  float    v2f;

#define BS    4
#define NN    2048
#define CIN   128
#define COUT  64
#define NH    4
#define HD    (COUT*NH)      // 256
#define LEAKY 0.1f
#define MASKF (-1e18f)
#define MASKTH (-1e17f)      // threshold to recover mask from score value

// ---------------------------------------------------------------------------
// Kernel A: hs = xs @ W1 with full-precision V_WMMA_F32_16X16X4_F32, plus
// ei = hs·a_l, ej = hs·a_r (in-wave butterfly reductions, no atomics), and
// hs stored transposed [b][h][d][n] in f16 so kernel B's WMMA B-fragments
// are single contiguous 32-byte loads.
// Grid: bs*(n/16) blocks x 128 threads (4 wave32 = 4 heads).
// ---------------------------------------------------------------------------
__global__ __launch_bounds__(128)
void gat_gemm_kernel(const float* __restrict__ xs, const float* __restrict__ W1,
                     const float* __restrict__ a_l, const float* __restrict__ a_r,
                     _Float16* __restrict__ hsT, float* __restrict__ ei,
                     float* __restrict__ ej)
{
    const int lane = threadIdx.x & 31;
    const int h    = threadIdx.x >> 5;          // head = wave id
    const int b    = blockIdx.x >> 7;           // n/16 = 128 row-tiles
    const int i0   = (blockIdx.x & 127) << 4;
    const int r    = lane & 15;
    const int hi   = lane >> 4;

    v8f c[4];
    #pragma unroll
    for (int t = 0; t < 4; ++t) c[t] = (v8f)0.f;

    const float* arow = xs + ((size_t)b * NN + (i0 + r)) * CIN;

#if __has_builtin(__builtin_amdgcn_wmma_f32_16x16x4_f32)
    // Full fp32 precision: V_WMMA_F32_16X16X4_F32, K stepped by 4.
    for (int k0 = 0; k0 < CIN; k0 += 4) {
        const int ka = k0 + 2 * hi;             // A 32-bit layout: lanes16-31 hold K=2,3
        v2f a;  a.x = arow[ka];  a.y = arow[ka + 1];
        #pragma unroll
        for (int t = 0; t < 4; ++t) {
            const int n0 = h * COUT + t * 16 + r;
            v2f bb; bb.x = W1[(size_t)ka * HD + n0];
                    bb.y = W1[(size_t)(ka + 1) * HD + n0];
            c[t] = __builtin_amdgcn_wmma_f32_16x16x4_f32(
                       false, a, false, bb, (short)0, c[t], false, false);
        }
    }
#else
    // Fallback: f16 inputs, f32 accumulate, K stepped by 32.
    for (int k0 = 0; k0 < CIN; k0 += 32) {
        const int kb = k0 + hi * 8;
        v16h a;
        #pragma unroll
        for (int q = 0; q < 8; ++q) {
            a[q]     = (_Float16)arow[kb + q];
            a[8 + q] = (_Float16)arow[kb + 16 + q];
        }
        #pragma unroll
        for (int t = 0; t < 4; ++t) {
            const int n0 = h * COUT + t * 16 + r;
            const int kc = k0 + hi * 16;
            v16h bb;
            #pragma unroll
            for (int q = 0; q < 16; ++q)
                bb[q] = (_Float16)W1[(size_t)(kc + q) * HD + n0];
            c[t] = __builtin_amdgcn_wmma_f32_16x16x32_f16(
                       false, a, false, bb, (short)0, c[t], false, false);
        }
    }
#endif

    // ei/ej: dot each hs row with a_l / a_r; butterfly-reduce the 16 lanes of
    // each half (columns), leaving the row sums in lanes 0 and 16.
    float sl[8], sr[8];
    #pragma unroll
    for (int v = 0; v < 8; ++v) { sl[v] = 0.f; sr[v] = 0.f; }
    #pragma unroll
    for (int t = 0; t < 4; ++t) {
        const float al = a_l[t * 16 + r];
        const float ar = a_r[t * 16 + r];
        #pragma unroll
        for (int v = 0; v < 8; ++v) {
            sl[v] += c[t][v] * al;
            sr[v] += c[t][v] * ar;
        }
    }
    #pragma unroll
    for (int off = 1; off < 16; off <<= 1) {
        #pragma unroll
        for (int v = 0; v < 8; ++v) {
            sl[v] += __shfl_xor(sl[v], off, 32);
            sr[v] += __shfl_xor(sr[v], off, 32);
        }
    }
    if (r == 0) {   // lane 0 holds rows 0..7, lane 16 rows 8..15
        const size_t base = ((size_t)(b * NH + h)) * NN + i0 + 8 * hi;
        #pragma unroll
        for (int v = 0; v < 8; ++v) { ei[base + v] = sl[v]; ej[base + v] = sr[v]; }
    }

    // hs transposed [b][h][d][n], f16. C layout: row = v + 8*hi, col = 16t + r.
    #pragma unroll
    for (int t = 0; t < 4; ++t) {
        const int d = t * 16 + r;
        const size_t rowb = ((size_t)(b * NH + h) * COUT + d) * NN + i0 + 8 * hi;
        #pragma unroll
        for (int v = 0; v < 8; ++v) hsT[rowb + v] = (_Float16)c[t][v];
    }
}

// ---------------------------------------------------------------------------
// Kernel B: fused masked-softmax attention, flash style — never materializes
// the 256 MB attention matrix. Each wave handles one head for a 16-row tile,
// streaming j in chunks of 32; scores are rank-1 (ei[i]+ej[j]) so the only
// matmul is P@hs, done with v_wmma_f32_16x16x32_f16 (f32 accumulate).
// Cross-half / cross-row broadcasts use ds_bpermute (__shfl) only — no LDS,
// no DS-counter drains.
// ---------------------------------------------------------------------------
__global__ __launch_bounds__(128)
void gat_attn_kernel(const float* __restrict__ adjs,
                     const _Float16* __restrict__ hsT,
                     const float* __restrict__ ei, const float* __restrict__ ej,
                     float* __restrict__ out)
{
    const int lane = threadIdx.x & 31;
    const int h    = threadIdx.x >> 5;
    const int b    = blockIdx.x >> 7;
    const int i0   = (blockIdx.x & 127) << 4;
    const int r    = lane & 15;          // score row within tile (A-frag row)
    const int hi   = lane >> 4;
    const int kb   = hi * 16;            // A 16-bit layout K-base per lane half (contiguous)

    const float  ei_own   = ei[((size_t)(b * NH + h)) * NN + i0 + r];
    const float* ejrow    = ej + ((size_t)(b * NH + h)) * NN;
    const float* adjrow   = adjs + ((size_t)b * NN + (i0 + r)) * NN;
    const _Float16* hbase = hsT + ((size_t)(b * NH + h) * COUT) * NN;

    float m_own = -3.0e38f;   // running row max (replicated in both lane halves)
    float l_own = 0.f;        // running row denominator
    v8f c[4];
    #pragma unroll
    for (int t = 0; t < 4; ++t) c[t] = (v8f)0.f;

    for (int j0 = 0; j0 < NN; j0 += 32) {
        if (j0 + 256 < NN) __builtin_prefetch(adjrow + j0 + 256, 0, 1);

        // This lane's 16 j-positions in A-fragment K order: {kb..kb+15}.
        float av[16], e[16];
        *(float4*)(av + 0)  = *(const float4*)(adjrow + j0 + kb);
        *(float4*)(av + 4)  = *(const float4*)(adjrow + j0 + kb + 4);
        *(float4*)(av + 8)  = *(const float4*)(adjrow + j0 + kb + 8);
        *(float4*)(av + 12) = *(const float4*)(adjrow + j0 + kb + 12);
        *(float4*)(e + 0)   = *(const float4*)(ejrow + j0 + kb);
        *(float4*)(e + 4)   = *(const float4*)(ejrow + j0 + kb + 4);
        *(float4*)(e + 8)   = *(const float4*)(ejrow + j0 + kb + 8);
        *(float4*)(e + 12)  = *(const float4*)(ejrow + j0 + kb + 12);

        // Masked leaky-relu scores, computed in place; chunk max.
        float cmax = -3.0e38f;
        #pragma unroll
        for (int q = 0; q < 16; ++q) {
            float x = ei_own + e[q];
            x = (x > 0.f) ? x : LEAKY * x;
            x = (av[q] != 0.f) ? x : MASKF;   // av dead after this
            e[q] = x;
            cmax = fmaxf(cmax, x);
        }
        cmax = fmaxf(cmax, __shfl_xor(cmax, 16, 32));  // join the two K halves

        const float m_new = fmaxf(m_own, cmax);
        const float scale = __expf(m_own - m_new);

        // Denominator counts masked exp terms (matches reference softmax);
        // numerator folds in the adj multiply (mask <=> e == MASKF).
        float csum = 0.f;
        v16h a;
        #pragma unroll
        for (int q = 0; q < 16; ++q) {
            const float p = __expf(e[q] - m_new);
            csum += p;
            a[q] = (_Float16)((e[q] > MASKTH) ? p : 0.f);
        }
        csum += __shfl_xor(csum, 16, 32);
        l_own = l_own * scale + csum;
        m_own = m_new;

        // Per-row rescale of the accumulator: C row v+8*hi's scale lives in
        // lane v+8*hi — broadcast via ds_bpermute, no LDS, no waits.
        #pragma unroll
        for (int v = 0; v < 8; ++v) {
            const float scv = __shfl(scale, v + 8 * hi, 32);
            #pragma unroll
            for (int t = 0; t < 4; ++t) c[t][v] *= scv;
        }

        // Accumulate P(16x32) @ hs_chunk(32x64) as 4 WMMAs. B (like A) uses
        // contiguous K order: lane half holds K = 16*hi..16*hi+15 — contiguous in hsT's
        // [d][n] layout, single 32-byte load.
        #pragma unroll
        for (int t = 0; t < 4; ++t) {
            const _Float16* hp = hbase + ((size_t)(t * 16 + r)) * NN + j0 + hi * 16;
            v16h bb = *(const v16h*)hp;
            c[t] = __builtin_amdgcn_wmma_f32_16x16x32_f16(
                       false, a, false, bb, (short)0, c[t], false, false);
        }
    }

    // Normalize by row denominators (broadcast via shuffle) and store
    // out[b][i][h*64 + d].
    #pragma unroll
    for (int v = 0; v < 8; ++v) {
        const float lv = 1.f / __shfl(l_own, v + 8 * hi, 32);
        const size_t row = (size_t)b * NN + i0 + v + 8 * hi;
        #pragma unroll
        for (int t = 0; t < 4; ++t)
            out[row * HD + h * COUT + t * 16 + r] = c[t][v] * lv;
    }
}

// ---------------------------------------------------------------------------
extern "C" void kernel_launch(void* const* d_in, const int* in_sizes, int n_in,
                              void* d_out, int out_size, void* d_ws, size_t ws_size,
                              hipStream_t stream)
{
    const float* xs   = (const float*)d_in[0];
    const float* adjs = (const float*)d_in[1];
    const float* W1   = (const float*)d_in[2];
    const float* a_l  = (const float*)d_in[3];
    const float* a_r  = (const float*)d_in[4];
    float* out = (float*)d_out;

    char* ws = (char*)d_ws;
    _Float16* hsT = (_Float16*)ws;                                   // 4 MB
    float* ei = (float*)(ws + (size_t)BS * NH * COUT * NN * sizeof(_Float16));
    float* ej = ei + (size_t)BS * NH * NN;                           // 128 KB each

    const dim3 grid(BS * (NN / 16));   // 512 blocks
    const dim3 block(128);             // 4 wave32 = 4 heads

    gat_gemm_kernel<<<grid, block, 0, stream>>>(xs, W1, a_l, a_r, hsT, ei, ej);
    gat_attn_kernel<<<grid, block, 0, stream>>>(adjs, hsT, ei, ej, out);
}
